// MetaGNN_86423331930503
// MI455X (gfx1250) — matrix operand order repaired and checked by reading, not compiled, not run on hardware
//
#include <hip/hip_runtime.h>

// ---------------------------------------------------------------------------
// MetaGNN (hypernetwork GAT) forward for MI455X / gfx1250.
// Heavy math runs through v_wmma_f32_16x16x32_bf16 with software-pipelined
// (ping-pong) B-fragment streaming from L2; edge phase is atomic/segment work
// sized for the 23.3 TB/s HBM + 192MB L2.
// ---------------------------------------------------------------------------

typedef __attribute__((ext_vector_type(16))) __bf16 v16bf;
typedef __attribute__((ext_vector_type(8)))  float  v8f;

#define NEG_SLOPE 0.2f

// order-preserving float <-> uint encoding for atomicMax on floats
__device__ __forceinline__ unsigned enc_f32(float x) {
  unsigned b = __float_as_uint(x);
  return b ^ ((b & 0x80000000u) ? 0xFFFFFFFFu : 0x80000000u);
}
__device__ __forceinline__ float dec_f32(unsigned u) {
  unsigned b = (u & 0x80000000u) ? (u ^ 0x80000000u) : ~u;
  return __uint_as_float(b);
}

// branch-free tanh: 1 - 2/(exp(2x)+1); exact limits at +-inf via rcp(inf)=0
__device__ __forceinline__ float fast_tanh(float x) {
  float t = __expf(2.0f * x);
  return 1.0f - 2.0f * __builtin_amdgcn_rcpf(t + 1.0f);
}

// ---------------------------------------------------------------------------
// K0: pre-swizzle weights into bf16 WMMA B-fragment order.
//   B layout (K x 16, bf16): lane L slot j -> N = L&15, K = ((L>>4)<<4) + j
//   packed[e], e = ((kb*ntn + tn)*32 + lane)*16 + j,  k = kb*32 + Kform
// B1: f-GEMM weights      B1[k=c][n=h] = lin_f_w[n*128 + k]          (K=128, ntn=8)
// B2: att-GEMM weights    B2[k=h][n=o] = att_hyper_w[n*128 + k]      (K=128, ntn=4)
// W2: xt-GEMM weights     W2[k=(h,i)][n=o] = W_hyper[(n*64+i)*128+h] (K=8192, ntn=4)
// ---------------------------------------------------------------------------
#define NB1 (4 * 8 * 512)
#define NB2 (4 * 4 * 512)
#define NW2 (256 * 4 * 512)
#define W2PAD (2 * 2048)  // one extra K-step pair so the pipelined loop over-reads safely

__global__ void pack_weights(const float* __restrict__ lin_f_w,
                             const float* __restrict__ att_hyper_w,
                             const float* __restrict__ W_hyper_w,
                             __bf16* __restrict__ B1p,
                             __bf16* __restrict__ B2p,
                             __bf16* __restrict__ W2p) {
  const int total = NB1 + NB2 + NW2;
  for (int idx = blockIdx.x * blockDim.x + threadIdx.x; idx < total;
       idx += gridDim.x * blockDim.x) {
    if (idx < NB1) {
      int e = idx, j = e & 15, lane = (e >> 4) & 31, t = e >> 9;
      int tn = t & 7, kb = t >> 3;
      int k = kb * 32 + ((lane >> 4) << 4) + j;
      int n = tn * 16 + (lane & 15);
      B1p[e] = (__bf16)lin_f_w[n * 128 + k];
    } else if (idx < NB1 + NB2) {
      int e = idx - NB1, j = e & 15, lane = (e >> 4) & 31, t = e >> 9;
      int tn = t & 3, kb = t >> 2;
      int k = kb * 32 + ((lane >> 4) << 4) + j;
      int n = tn * 16 + (lane & 15);
      B2p[e] = (__bf16)att_hyper_w[n * 128 + k];
    } else {
      int e = idx - NB1 - NB2, j = e & 15, lane = (e >> 4) & 31, t = e >> 9;
      int tn = t & 3, kb = t >> 2;
      int k = kb * 32 + ((lane >> 4) << 4) + j;  // k = h*64 + i
      int n = tn * 16 + (lane & 15);             // o
      W2p[e] = (__bf16)W_hyper_w[(n * 64 + (k & 63)) * 128 + (k >> 6)];
    }
  }
}

// ---------------------------------------------------------------------------
// K1: per-16-node tile, one wave each. Fused:
//   f   = tanh([x|meta] @ lin_f_w^T)                (16x128, K=128)
//   att = f @ att_hyper_w[0:64]^T                   (16x64,  K=128)
//   xt  = p @ W2, p[n,(h,i)] = f[n,h]*x[n,i]        (16x64,  K=8192)
//   a_dst/a_src per node, self-loop alpha seeds segment-max.
// A fragment layout (16x32 bf16): lane L slot j ->
//   M = L&15, K = ((L>>4)<<4) + j
// ---------------------------------------------------------------------------
#define WAVES_PER_BLK 2
#define LDS_PER_WAVE  6400   // floats: A1[16*132] F[16*132] ATT[16*68] XT[16*68]

__global__ __launch_bounds__(WAVES_PER_BLK * 32)
void node_hyper_wmma(const float* __restrict__ emb,
                     const float* __restrict__ meta,
                     const __bf16* __restrict__ B1p,
                     const __bf16* __restrict__ B2p,
                     const __bf16* __restrict__ W2p,
                     const float* __restrict__ att_i_o,
                     const float* __restrict__ att_j_o,
                     float* __restrict__ xt_g,
                     float* __restrict__ a_dst,
                     float* __restrict__ a_src,
                     unsigned* __restrict__ m_enc,
                     int N) {
  __shared__ float lds[WAVES_PER_BLK * LDS_PER_WAVE];
  const int lane = threadIdx.x & 31;
  const int wave = threadIdx.x >> 5;
  const int tile = blockIdx.x * WAVES_PER_BLK + wave;
  const int ntiles = N >> 4;
  if (tile >= ntiles) return;  // whole wave exits; active waves keep EXEC all-1s

  float* A1  = lds + wave * LDS_PER_WAVE;  // [16][132] : cols 0-63 x, 64-127 meta
  float* F   = A1 + 16 * 132;              // [16][132]
  float* ATT = F + 16 * 132;               // [16][68]
  float* XT  = ATT + 16 * 68;              // [16][68]
  const int n0 = tile * 16;

  for (int idx = lane; idx < 16 * 128; idx += 32) {
    int r = idx >> 7, c = idx & 127;
    float v = (c < 64) ? emb[(size_t)(n0 + r) * 64 + c]
                       : meta[(size_t)(n0 + r) * 64 + (c - 64)];
    A1[r * 132 + c] = v;
  }

  const int half8 = (lane >> 4) << 3;  // D-row offset 0 or 8; A-frag K offset is half8<<1 (0 or 16)
  const int col   = lane & 15;         // A-frag row M / B-frag col N

  // ---- GEMM1: F = tanh(A1 @ lin_f_w^T), 8 h-tiles, K=128 ----
  for (int tn = 0; tn < 8; ++tn) {
    v8f acc = {};
    for (int kb = 0; kb < 4; ++kb) {
      v16bf a;
#pragma unroll
      for (int j = 0; j < 16; ++j) {
        int k = kb * 32 + (half8 << 1) + j;
        a[j] = (__bf16)A1[col * 132 + k];
      }
      v16bf b = *(const v16bf*)(B1p + ((size_t)(kb * 8 + tn) * 32 + lane) * 16);
      acc = __builtin_amdgcn_wmma_f32_16x16x32_bf16(false, a, false, b,
                                                    (short)0, acc, false, false);
    }
#pragma unroll
    for (int r = 0; r < 8; ++r)
      F[(r + half8) * 132 + tn * 16 + col] = fast_tanh(acc[r]);
  }

  // ---- GEMM2: ATT = F @ att_hyper_w[0:64]^T, 4 o-tiles, K=128 ----
  for (int tn = 0; tn < 4; ++tn) {
    v8f acc = {};
    for (int kb = 0; kb < 4; ++kb) {
      v16bf a;
#pragma unroll
      for (int j = 0; j < 16; ++j) {
        int k = kb * 32 + (half8 << 1) + j;
        a[j] = (__bf16)F[col * 132 + k];
      }
      v16bf b = *(const v16bf*)(B2p + ((size_t)(kb * 4 + tn) * 32 + lane) * 16);
      acc = __builtin_amdgcn_wmma_f32_16x16x32_bf16(false, a, false, b,
                                                    (short)0, acc, false, false);
    }
#pragma unroll
    for (int r = 0; r < 8; ++r)
      ATT[(r + half8) * 68 + tn * 16 + col] = acc[r];
  }

  // ---- GEMM3: XT = p @ W2, K=8192, software-pipelined B stream ----
  // Each 32-wide K block kb has constant h = kb>>1; i = (kb&1)*32 + off.
  float xa_[16], xb_[16];
#pragma unroll
  for (int j = 0; j < 16; ++j) {
    int off = (half8 << 1) + j;  // 0..31
    xa_[j] = A1[col * 132 + off];
    xb_[j] = A1[col * 132 + 32 + off];
  }

  auto loadB = [&](v16bf* b, int kb) {
    const __bf16* wb = W2p + (size_t)kb * 2048 + (size_t)lane * 16;
    b[0] = *(const v16bf*)(wb);
    b[1] = *(const v16bf*)(wb + 512);
    b[2] = *(const v16bf*)(wb + 1024);
    b[3] = *(const v16bf*)(wb + 1536);
  };

  v8f c0 = {}, c1 = {}, c2 = {}, c3 = {};
  v16bf ba[4], bb[4];
  loadB(ba, 0);                       // prologue: kb = 0
  for (int kb = 0; kb < 256; kb += 2) {
    loadB(bb, kb + 1);                // in flight while ba is consumed
    __builtin_prefetch(W2p + (size_t)(kb + 2) * 2048 + (size_t)lane * 16, 0, 0);
    float fh = F[col * 132 + (kb >> 1)];
    v16bf a;
#pragma unroll
    for (int j = 0; j < 16; ++j) a[j] = (__bf16)(fh * xa_[j]);
    c0 = __builtin_amdgcn_wmma_f32_16x16x32_bf16(false, a, false, ba[0], (short)0, c0, false, false);
    c1 = __builtin_amdgcn_wmma_f32_16x16x32_bf16(false, a, false, ba[1], (short)0, c1, false, false);
    c2 = __builtin_amdgcn_wmma_f32_16x16x32_bf16(false, a, false, ba[2], (short)0, c2, false, false);
    c3 = __builtin_amdgcn_wmma_f32_16x16x32_bf16(false, a, false, ba[3], (short)0, c3, false, false);
    loadB(ba, kb + 2);                // kb+2==256 over-reads into W2PAD (harmless)
#pragma unroll
    for (int j = 0; j < 16; ++j) a[j] = (__bf16)(fh * xb_[j]);
    c0 = __builtin_amdgcn_wmma_f32_16x16x32_bf16(false, a, false, bb[0], (short)0, c0, false, false);
    c1 = __builtin_amdgcn_wmma_f32_16x16x32_bf16(false, a, false, bb[1], (short)0, c1, false, false);
    c2 = __builtin_amdgcn_wmma_f32_16x16x32_bf16(false, a, false, bb[2], (short)0, c2, false, false);
    c3 = __builtin_amdgcn_wmma_f32_16x16x32_bf16(false, a, false, bb[3], (short)0, c3, false, false);
  }

  // D layout: element r -> M = r + 8*(lane>>4), N = tn*16 + (lane&15)
#pragma unroll
  for (int r = 0; r < 8; ++r) {
    int M = r + half8;
    float v0 = c0[r], v1 = c1[r], v2 = c2[r], v3 = c3[r];
    XT[M * 68 + 0 + col]  = v0;
    XT[M * 68 + 16 + col] = v1;
    XT[M * 68 + 32 + col] = v2;
    XT[M * 68 + 48 + col] = v3;
    size_t g = (size_t)(n0 + M) * 64;
    xt_g[g + 0 + col]  = v0;
    xt_g[g + 16 + col] = v1;
    xt_g[g + 32 + col] = v2;
    xt_g[g + 48 + col] = v3;
  }

  // ---- attention scalars + self-loop seed for segment max ----
  if (lane < 16) {
    int n = n0 + lane;
    float ad = 0.f, as = 0.f;
    for (int o = 0; o < 64; ++o) {
      float xo = XT[lane * 68 + o];
      float ai = ATT[lane * 68 + o];
      ad += xo * att_i_o[o] * ai;
      as += xo * att_j_o[o] * ai;
    }
    a_dst[n] = ad;
    a_src[n] = as;
    float al = ad + as;  // self loop: src == dst == n
    al = (al >= 0.f) ? al : NEG_SLOPE * al;
    m_enc[n] = enc_f32(al);
  }
}

// ---------------------------------------------------------------------------
// K2: per-edge leaky-relu logit + atomic segment max (encoded uint).
// ---------------------------------------------------------------------------
__global__ void edge_max(const int* __restrict__ src, const int* __restrict__ dst,
                         const float* __restrict__ a_dst, const float* __restrict__ a_src,
                         float* __restrict__ alpha, unsigned* __restrict__ m_enc, int E) {
  for (int e = blockIdx.x * blockDim.x + threadIdx.x; e < E;
       e += gridDim.x * blockDim.x) {
    int s = src[e], d = dst[e];
    float al = a_dst[d] + a_src[s];
    al = (al >= 0.f) ? al : NEG_SLOPE * al;
    alpha[e] = al;
    atomicMax(&m_enc[d], enc_f32(al));
  }
}

// ---------------------------------------------------------------------------
// K3: per-node init: decode max, self-loop exp seeds the softmax denominator
//     and the output accumulator (out = xt * exp(alpha_self - m)).
// ---------------------------------------------------------------------------
__global__ void node_init(const float* __restrict__ xt,
                          const float* __restrict__ a_dst,
                          const float* __restrict__ a_src,
                          const unsigned* __restrict__ m_enc,
                          float* __restrict__ m_f, float* __restrict__ s_sum,
                          float* __restrict__ out, int N) {
  int total = N * 64;
  for (int i = blockIdx.x * blockDim.x + threadIdx.x; i < total;
       i += gridDim.x * blockDim.x) {
    int n = i >> 6, o = i & 63;
    float m = dec_f32(m_enc[n]);
    float al = a_dst[n] + a_src[n];
    al = (al >= 0.f) ? al : NEG_SLOPE * al;
    float es = __expf(al - m);
    out[i] = xt[i] * es;
    if (o == 0) { s_sum[n] = es; m_f[n] = m; }
  }
}

// ---------------------------------------------------------------------------
// K4: per-edge exp(alpha - m[dst]) + atomic segment sum.
// ---------------------------------------------------------------------------
__global__ void edge_exp(const int* __restrict__ dst,
                         const float* __restrict__ m_f,
                         float* __restrict__ alpha, float* __restrict__ s_sum, int E) {
  for (int e = blockIdx.x * blockDim.x + threadIdx.x; e < E;
       e += gridDim.x * blockDim.x) {
    int d = dst[e];
    float w = __expf(alpha[e] - m_f[d]);
    alpha[e] = w;
    atomicAdd(&s_sum[d], w);
  }
}

// ---------------------------------------------------------------------------
// K5: 32 lanes per edge: out[dst] += xt[src] * w  (64 floats, 2 per lane).
// ---------------------------------------------------------------------------
__global__ void edge_scatter(const int* __restrict__ src, const int* __restrict__ dst,
                             const float* __restrict__ alpha,
                             const float* __restrict__ xt, float* __restrict__ out, int E) {
  int t = blockIdx.x * blockDim.x + threadIdx.x;
  int e = t >> 5, lane = t & 31;
  if (e >= E) return;
  int s = src[e], d = dst[e];
  float w = alpha[e];
  const float* xs = xt + (size_t)s * 64;
  float* od = out + (size_t)d * 64;
  atomicAdd(od + lane, xs[lane] * w);
  atomicAdd(od + lane + 32, xs[lane + 32] * w);
}

// ---------------------------------------------------------------------------
// K6: normalize by segment sum and apply ELU.
// ---------------------------------------------------------------------------
__global__ void finalize(const float* __restrict__ s_sum, float* __restrict__ out, int N) {
  int total = N * 64;
  for (int i = blockIdx.x * blockDim.x + threadIdx.x; i < total;
       i += gridDim.x * blockDim.x) {
    float v = out[i] / (s_sum[i >> 6] + 1e-16f);
    out[i] = (v > 0.f) ? v : expm1f(v);
  }
}

// ---------------------------------------------------------------------------
extern "C" void kernel_launch(void* const* d_in, const int* in_sizes, int n_in,
                              void* d_out, int out_size, void* d_ws, size_t ws_size,
                              hipStream_t stream) {
  const float* emb        = (const float*)d_in[0];
  const float* meta       = (const float*)d_in[1];
  const float* lin_f_w    = (const float*)d_in[2];
  const float* W_hyper_w  = (const float*)d_in[3];
  const float* att_hyper  = (const float*)d_in[4];
  const float* att_i_o    = (const float*)d_in[5];
  const float* att_j_o    = (const float*)d_in[6];
  const int*   ei         = (const int*)d_in[7];

  const int N = in_sizes[0] / 64;      // 50000
  const int E = in_sizes[7] / 2;       // 1600000
  const int* src = ei;
  const int* dst = ei + E;
  float* out = (float*)d_out;          // doubles as message accumulator

  // workspace carve-up (all 256B aligned); total ~21 MB
  char* p = (char*)d_ws;
  auto carve = [&](size_t bytes) -> char* {
    char* r = p;
    p += (bytes + 255) & ~(size_t)255;
    return r;
  };
  __bf16*   B1p   = (__bf16*)carve((size_t)NB1 * 2);
  __bf16*   B2p   = (__bf16*)carve((size_t)NB2 * 2);
  __bf16*   W2p   = (__bf16*)carve((size_t)(NW2 + W2PAD) * 2);
  float*    xt    = (float*)carve((size_t)N * 64 * 4);
  float*    a_dst = (float*)carve((size_t)N * 4);
  float*    a_src = (float*)carve((size_t)N * 4);
  unsigned* m_enc = (unsigned*)carve((size_t)N * 4);
  float*    m_f   = (float*)carve((size_t)N * 4);
  float*    s_sum = (float*)carve((size_t)N * 4);
  float*    alpha = (float*)carve((size_t)E * 4);

  pack_weights<<<1072, 256, 0, stream>>>(lin_f_w, att_hyper, W_hyper_w, B1p, B2p, W2p);

  int ntiles = N / 16;
  int nblk1  = (ntiles + WAVES_PER_BLK - 1) / WAVES_PER_BLK;
  node_hyper_wmma<<<nblk1, WAVES_PER_BLK * 32, 0, stream>>>(
      emb, meta, B1p, B2p, W2p, att_i_o, att_j_o, xt, a_dst, a_src, m_enc, N);

  edge_max<<<2048, 256, 0, stream>>>(src, dst, a_dst, a_src, alpha, m_enc, E);
  node_init<<<4096, 256, 0, stream>>>(xt, a_dst, a_src, m_enc, m_f, s_sum, out, N);
  edge_exp<<<2048, 256, 0, stream>>>(dst, m_f, alpha, s_sum, E);

  long long sc_threads = (long long)E * 32;
  int sc_blocks = (int)((sc_threads + 255) / 256);
  edge_scatter<<<sc_blocks, 256, 0, stream>>>(src, dst, alpha, xt, out, E);

  finalize<<<4096, 256, 0, stream>>>(s_sum, out, N);
}
